// KoLeoLoss_19292993093763
// MI455X (gfx1250) — matrix operand order repaired and checked by reading, not compiled, not run on hardware
//
#include <hip/hip_runtime.h>
#include <hip/hip_bf16.h>
#include <math.h>

// ---------------------------------------------------------------------------
// KoLeo loss for MI455X (gfx1250, wave32, WMMA).
//   x  : [8192, 768] fp32 -> scalar fp32 loss
// k_normalize   : L2-normalize rows -> xn (fp32) + xb (bf16)
// k_gemm_argmax : dots = xb @ xb^T via v_wmma_f32_16x16x32_bf16.
//                 Grid = 64 row-blocks x 8 N-chunks (512 blocks) to fill the
//                 chip. B panels double-buffered in LDS, global->LDS staging
//                 pipelined under the WMMA chain, B fragments ping-pong
//                 double-buffered with sched_group_barrier-enforced
//                 (DS2)(DS2 WMMA)* interleave so every WMMA waits dscnt<=2.
//                 Per-row partial argmax kept as sortable u32 key
//                 (float-order bits | 8191-col), one key per (row, chunk).
// k_dist        : max-combines the 8 chunk keys per row (deterministic),
//                 then -log(||x_i - x_nn(i) + eps|| + eps)
// k_reduce      : mean -> d_out[0]
// ---------------------------------------------------------------------------

typedef __bf16 bf16_t;
typedef __attribute__((ext_vector_type(16))) __bf16 v16bf;
typedef __attribute__((ext_vector_type(8)))  __bf16 v8bf;
typedef __attribute__((ext_vector_type(8)))  float  v8f;

#define NROWS 8192
#define DIM   768
#define KTILES 24                  // 768 / 32
#define EPSF  1e-8f

#define NCHUNKS 8
#define NCHUNK  (NROWS / NCHUNKS)  // 1024 columns per chunk
#define TPC     (NCHUNK / 16)      // 64 tiles per chunk

// LDS B panel: 16 rows x 776 bf16 (768 + 8 pad -> 388-dword stride, so each
// 16-lane half of a ds_load_b128 covers all 64 banks exactly once).
#define BP_ROW_B   1552            // bytes per padded row
#define BP_BYTES   (16 * BP_ROW_B) // 24832 B per panel

#if __has_builtin(__builtin_amdgcn_sched_group_barrier)
#define SGB(mask, size, id) __builtin_amdgcn_sched_group_barrier(mask, size, id)
#else
#define SGB(mask, size, id)
#endif
#define SGB_DSREAD 0x100
#define SGB_WMMA   0x008

// ---------------- kernel 1: row L2-normalize --------------------------------
__global__ void __launch_bounds__(256)
k_normalize(const float* __restrict__ x,
            float* __restrict__ xn,
            bf16_t* __restrict__ xb) {
  const int row = blockIdx.x;
  const float* r = x + (size_t)row * DIM;
  float v[3];
  float s = 0.f;
#pragma unroll
  for (int i = 0; i < 3; ++i) {
    v[i] = r[threadIdx.x + i * 256];
    s += v[i] * v[i];
  }
#pragma unroll
  for (int off = 16; off; off >>= 1) s += __shfl_xor(s, off, 32);
  __shared__ float red[8];
  if ((threadIdx.x & 31) == 0) red[threadIdx.x >> 5] = s;
  __syncthreads();
  float tot = 0.f;
#pragma unroll
  for (int i = 0; i < 8; ++i) tot += red[i];
  const float inv = 1.f / fmaxf(sqrtf(tot), EPSF);
#pragma unroll
  for (int i = 0; i < 3; ++i) {
    const float nv = v[i] * inv;
    const size_t o = (size_t)row * DIM + threadIdx.x + i * 256;
    xn[o] = nv;
    xb[o] = (bf16_t)nv;
  }
}

// ---- A fragment from global (ISA 7.12.2, 16-bit A 16x32) -------------------
__device__ __forceinline__ v16bf load_fragA(const bf16_t* __restrict__ xb,
                                            int row0, int kb) {
  const int lane = threadIdx.x & 31;
  const int half = lane >> 4;
  const int r    = lane & 15;
  const bf16_t* p = xb + (size_t)(row0 + r) * DIM + kb + half * 8;
  union { v16bf f; v8bf h[2]; } u;
  u.h[0] = *(const v8bf*)(p);
  u.h[1] = *(const v8bf*)(p + 16);
  return u.f;
}

// ---------------- kernel 2: WMMA GEMM + streamed partial argmax -------------
__global__ void __launch_bounds__(256)
k_gemm_argmax(const bf16_t* __restrict__ xb, unsigned int* __restrict__ keys) {
  const int wave  = threadIdx.x >> 5;
  const int lane  = threadIdx.x & 31;
  const int half  = lane >> 4;
  const int r     = lane & 15;
  const int m0    = blockIdx.x * 128 + wave * 16;
  const int chunk = blockIdx.y;
  const int nbase = chunk * NCHUNK;

  __shared__ __align__(16) char bpan[2][BP_BYTES];   // double-buffered panels

  // staging geometry: thread t copies 96 B of the 24 KB panel (3 x 2 b128)
  const int srow   = threadIdx.x >> 4;   // 0..15
  const int schunk = threadIdx.x & 15;   // 0..15

  // register-resident A fragments: read once, reused for all 64 N tiles
  v16bf aF[KTILES];
#pragma unroll
  for (int k = 0; k < KTILES; ++k) aF[k] = load_fragA(xb, m0, k * 32);

  // fused (value, argmax) key per row: top 19 bits = order-isomorphic float
  // bits, low 13 bits = 8191 - col  (u32 max == argmax with lowest-col ties)
  unsigned int bk[8];
#pragma unroll
  for (int j = 0; j < 8; ++j) bk[j] = 0u;

  // ---- prologue: stage panel for n0 = nbase ----
  {
    const bf16_t* gp = xb + (size_t)(nbase + srow) * DIM + schunk * 48;
    char*         lp = bpan[0] + srow * BP_ROW_B + schunk * 96;
#pragma unroll
    for (int ph = 0; ph < 3; ++ph) {
      uint4 g0 = *(const uint4*)(gp + ph * 16);
      uint4 g1 = *(const uint4*)(gp + ph * 16 + 8);
      *(uint4*)(lp + ph * 32)      = g0;
      *(uint4*)(lp + ph * 32 + 16) = g1;
    }
  }
  __syncthreads();

  for (int t = 0; t < TPC; ++t) {
    const int  n0   = nbase + t * 16;
    const int  cur  = t & 1;
    const bool more = (t + 1) < TPC;
    const char* bcur = bpan[cur];
    char*       bnxt = bpan[cur ^ 1];
    const bf16_t* gp = xb + (size_t)(n0 + 16 + srow) * DIM + schunk * 48;
    char*         lp = bnxt + srow * BP_ROW_B + schunk * 96;

    uint4 g0, g1;
    if (more) {                    // staging phase 0 in flight over wmma 0..7
      g0 = *(const uint4*)(gp);
      g1 = *(const uint4*)(gp + 8);
    }

    // Ping-pong double buffer of B fragments; sched_group_barrier enforces
    // (DS2)(DS2)(W)(DS2)(W)... so loads for step k+1 schedule before WMMA k
    // and every WMMA waits only dscnt<=2.
    union BF { v16bf f; v8bf h[2]; };
    BF ue, uo;
    {
      const char* p = bcur + r * BP_ROW_B + (half * 8) * 2;
      ue.h[0] = *(const v8bf*)(p);
      ue.h[1] = *(const v8bf*)(p + 32);
    }
    SGB(SGB_DSREAD, 2, 0);         // group: initial ue loads

    v8f c = {};
#pragma unroll
    for (int k = 0; k < KTILES; k += 2) {
      {
        const char* p = bcur + r * BP_ROW_B + ((k + 1) * 32 + half * 8) * 2;
        uo.h[0] = *(const v8bf*)(p);
        uo.h[1] = *(const v8bf*)(p + 32);
      }
      SGB(SGB_DSREAD, 2, 0);       // group: loads for k+1
      SGB(SGB_WMMA,   1, 0);       // group: WMMA k
      c = __builtin_amdgcn_wmma_f32_16x16x32_bf16(
              false, aF[k], false, ue.f, (short)0, c, false, false);

      if (k + 2 < KTILES) {
        const char* p = bcur + r * BP_ROW_B + ((k + 2) * 32 + half * 8) * 2;
        ue.h[0] = *(const v8bf*)(p);
        ue.h[1] = *(const v8bf*)(p + 32);
      }
      if (k == 6 && more) {        // retire staging phase 0, launch phase 1
        *(uint4*)(lp)      = g0;
        *(uint4*)(lp + 16) = g1;
        g0 = *(const uint4*)(gp + 16);
        g1 = *(const uint4*)(gp + 24);
      }
      SGB(SGB_DSREAD, 2, 0);       // group: loads for k+2
      SGB(SGB_WMMA,   1, 0);       // group: WMMA k+1
      c = __builtin_amdgcn_wmma_f32_16x16x32_bf16(
              false, aF[k + 1], false, uo.f, (short)0, c, false, false);

      if (k == 14 && more) {       // retire staging phase 1, launch phase 2
        *(uint4*)(lp + 32) = g0;
        *(uint4*)(lp + 48) = g1;
        g0 = *(const uint4*)(gp + 32);
        g1 = *(const uint4*)(gp + 40);
      }
    }
    if (more) {                    // retire staging phase 2
      *(uint4*)(lp + 64) = g0;
      *(uint4*)(lp + 80) = g1;
    }

    // C/D layout: N = lane&15, M = half*8 + vgpr-index
    const int col = n0 + r;
    const unsigned int colkey = 8191u ^ (unsigned int)col;
#pragma unroll
    for (int j = 0; j < 8; ++j) {
      const int rowg = m0 + half * 8 + j;
      const float v = (rowg == col) ? -1.0f : c[j];
      const unsigned int bits = __float_as_uint(v);
      const unsigned int srt  = bits ^ (unsigned int)(((int)bits >> 31) | 0x80000000);
      const unsigned int key  = (srt & 0xFFFFE000u) | colkey;
      bk[j] = (bk[j] > key) ? bk[j] : key;
    }
    __syncthreads();               // reads of bcur done; bnxt fully written
  }

  // reduce across the 16 lanes holding the same row (offsets stay in-half)
#pragma unroll
  for (int j = 0; j < 8; ++j) {
#pragma unroll
    for (int off = 1; off < 16; off <<= 1) {
      const unsigned int ok =
          (unsigned int)__shfl_xor((int)bk[j], off, 32);
      bk[j] = (bk[j] > ok) ? bk[j] : ok;
    }
  }
  if (r == 0) {
#pragma unroll
    for (int j = 0; j < 8; ++j)
      keys[(size_t)(m0 + half * 8 + j) * NCHUNKS + chunk] = bk[j];
  }
}

// ---------------- kernel 3: combine keys + distance + per-row loss ----------
__global__ void __launch_bounds__(256)
k_dist(const float* __restrict__ xn, const unsigned int* __restrict__ keys,
       float* __restrict__ partial) {
  const int row = blockIdx.x;
  unsigned int bestk = 0u;
#pragma unroll
  for (int i = 0; i < NCHUNKS; ++i) {
    const unsigned int k = keys[(size_t)row * NCHUNKS + i];
    bestk = (bestk > k) ? bestk : k;
  }
  const int nbr = (int)(8191u ^ (bestk & 8191u));

  const float* a = xn + (size_t)row * DIM;
  const float* b = xn + (size_t)nbr * DIM;
  float s = 0.f;
#pragma unroll
  for (int i = 0; i < 3; ++i) {
    const int c0 = threadIdx.x + i * 256;
    const float d = a[c0] - b[c0] + EPSF;
    s += d * d;
  }
#pragma unroll
  for (int off = 16; off; off >>= 1) s += __shfl_xor(s, off, 32);
  __shared__ float red[8];
  if ((threadIdx.x & 31) == 0) red[threadIdx.x >> 5] = s;
  __syncthreads();
  if (threadIdx.x == 0) {
    float tot = 0.f;
#pragma unroll
    for (int i = 0; i < 8; ++i) tot += red[i];
    partial[row] = -logf(sqrtf(tot) + EPSF);
  }
}

// ---------------- kernel 4: final mean --------------------------------------
__global__ void __launch_bounds__(256)
k_reduce(const float* __restrict__ partial, float* __restrict__ out) {
  float s = 0.f;
  for (int i = threadIdx.x; i < NROWS; i += 256) s += partial[i];
#pragma unroll
  for (int off = 16; off; off >>= 1) s += __shfl_xor(s, off, 32);
  __shared__ float red[8];
  if ((threadIdx.x & 31) == 0) red[threadIdx.x >> 5] = s;
  __syncthreads();
  if (threadIdx.x == 0) {
    float tot = 0.f;
#pragma unroll
    for (int i = 0; i < 8; ++i) tot += red[i];
    out[0] = tot / (float)NROWS;
  }
}

// ---------------- launch ----------------------------------------------------
extern "C" void kernel_launch(void* const* d_in, const int* in_sizes, int n_in,
                              void* d_out, int out_size, void* d_ws, size_t ws_size,
                              hipStream_t stream) {
  (void)in_sizes; (void)n_in; (void)out_size; (void)ws_size;
  const float* x = (const float*)d_in[0];
  char* ws = (char*)d_ws;

  const size_t xn_bytes = (size_t)NROWS * DIM * sizeof(float);    // 25165824
  const size_t xb_bytes = (size_t)NROWS * DIM * sizeof(bf16_t);   // 12582912
  const size_t ky_bytes = (size_t)NROWS * NCHUNKS * sizeof(unsigned int);

  float*        xn      = (float*)ws;
  bf16_t*       xb      = (bf16_t*)(ws + xn_bytes);
  unsigned int* keys    = (unsigned int*)(ws + xn_bytes + xb_bytes);
  float*        partial = (float*)(ws + xn_bytes + xb_bytes + ky_bytes);
  float*        out     = (float*)d_out;

  k_normalize  <<<NROWS, 256, 0, stream>>>(x, xn, xb);
  dim3 ggrid(NROWS / 128, NCHUNKS);
  k_gemm_argmax<<<ggrid, 256, 0, stream>>>(xb, keys);
  k_dist       <<<NROWS, 256, 0, stream>>>(xn, keys, partial);
  k_reduce     <<<1,     256, 0, stream>>>(partial, out);
}
